// SparseAttention_55138790146499
// MI455X (gfx1250) — matrix operand order repaired and checked by reading, not compiled, not run on hardware
//
#include <hip/hip_runtime.h>
#include <hip/hip_bf16.h>

typedef __attribute__((ext_vector_type(16))) _Float16 v16h;
typedef __attribute__((ext_vector_type(4)))  _Float16 v4h;
typedef __attribute__((ext_vector_type(8)))  float    v8f;
typedef __attribute__((ext_vector_type(4)))  float    v4f;
typedef __attribute__((ext_vector_type(4)))  unsigned int v4u;
typedef __attribute__((ext_vector_type(8)))  int      v8i;
typedef __attribute__((ext_vector_type(4)))  int      v4i;

#define B_    2
#define S_    2048
#define D_    1024
#define H_    16
#define DK_   64
#define M_    (B_ * S_)              // 4096
#define HEADSZ (S_ * DK_)            // 131072 elems per (b,h)
#define HELEMS (B_ * H_ * S_ * DK_)  // 4,194,304

// ---------------- WMMA fragment loaders (wave32, contiguous k order in A and B) ----------------
// A 16x32 f16: lane<16 -> K=0..15 ; lane>=16 -> K=16..31
__device__ __forceinline__ v16h load_a_frag(const _Float16* __restrict__ lds,
                                            int row0, int ld, int kbase, int lane) {
    int m = lane & 15;
    int h = lane >> 4;
    const _Float16* p0 = lds + (row0 + m) * ld + kbase + 16 * h;
    const _Float16* p1 = p0 + 8;
    v16h a;
#pragma unroll
    for (int i = 0; i < 8; ++i) a[i] = p0[i];
#pragma unroll
    for (int i = 0; i < 8; ++i) a[8 + i] = p1[i];
    return a;
}

// B 32x16 f16, LDS stored transposed as [n][k]: lane<16 K=0..15, lane>=16 K=16..31
__device__ __forceinline__ v16h load_bt_frag(const _Float16* __restrict__ lds,
                                             int n0, int ld, int kbase, int lane) {
    int n = lane & 15;
    int h = lane >> 4;
    const _Float16* p = lds + (n0 + n) * ld + kbase + 16 * h;
    v16h b;
#pragma unroll
    for (int i = 0; i < 16; ++i) b[i] = p[i];
    return b;
}

__device__ __forceinline__ v8f wmma_f16(v16h a, v16h b, v8f c) {
    return __builtin_amdgcn_wmma_f32_16x16x32_f16(false, a, false, b, (short)0, c, false, false);
}

// ---------------- Tensor Data Mover: 2D f16 tile -> LDS with HW row padding ----------------
// D# per CDNA5 ISA: group0 {count=1, lds_addr, global_addr, type=2},
// group1 {data_size=2B, pad_enable, pad_interval, pad_amount, tensor/tile dims, dim0 stride}.
// pad_interval code v -> 2^(v+1) DWORDs between pads; pad_amount code v -> v+1 DWORDs pad.
__device__ __forceinline__ void tdm_load_tile_f16(
    unsigned lds_off, const _Float16* gptr,
    unsigned width, unsigned rows, unsigned row_stride,
    unsigned pad_interval_code, unsigned pad_amount_code) {
    unsigned long long ga = (unsigned long long)gptr;
    v4u g0 = { 1u,                                   // count=1 (valid), user mode
               lds_off,                              // LDS byte address
               (unsigned)ga,                         // global_addr[31:0]
               ((unsigned)(ga >> 32) & 0x1FFFFFFu) | 0x80000000u };  // addr[56:32] | type=2
    v8i g1 = { (int)(0x10000u                        // data_size=1 (2 bytes)
                     | (1u << 20)                    // pad_enable
                     | (pad_interval_code << 22)
                     | (pad_amount_code << 25)),
               (int)(width << 16),                   // tensor_dim0 (bits 79:48, low part)
               (int)(rows << 16),                    // tensor_dim1 (bits 111:80, low part)
               (int)(width << 16),                   // tile_dim0   (bits 127:112)
               (int)rows,                            // tile_dim1   (bits 143:128)
               (int)row_stride,                      // tensor_dim0_stride[31:0]
               0, 0 };
    v4i z4 = {0, 0, 0, 0};
#if defined(__clang_major__) && (__clang_major__ >= 23)
    v8i z8 = {0, 0, 0, 0, 0, 0, 0, 0};
    __builtin_amdgcn_tensor_load_to_lds(g0, g1, z4, z4, z8, 0);
#else
    __builtin_amdgcn_tensor_load_to_lds(g0, g1, z4, z4, 0);
#endif
}

// ---------------- Kernel 1/4: double-buffered WMMA GEMM ----------------
// Y[4096,1024] = X[4096,1024] @ W[1024,1024] + bias
// MODE 0: X = f32 row-major,         out = f16 scattered to [B,H,S,dk]
// MODE 1: X = f16 ctx in [B,H,S,dk], out = f32 [B,S,D] (final output)
template <int MODE>
__global__ __launch_bounds__(256) void gemm_wmma(
    const float* __restrict__ Xf, const _Float16* __restrict__ Xh,
    const float* __restrict__ W, const float* __restrict__ bias,
    _Float16* __restrict__ Oh, float* __restrict__ Of) {

    __shared__ __align__(16) _Float16 sA[2][128 * 40];  // [row][k], ld=40
    __shared__ __align__(16) _Float16 sB[2][128 * 40];  // [n][k]  , ld=40 (transposed)

    const int t    = threadIdx.x;
    const int lane = t & 31;
    const int wave = t >> 5;
    const int wm   = wave >> 1;        // 0..3 -> 32-row band
    const int wn   = wave & 1;         // 0..1 -> 64-col band
    const int m0   = blockIdx.y * 128;
    const int n0   = blockIdx.x * 128;

    // --- tile fills: 16B global loads, 8B LDS stores ---
    auto fillA = [&](int buf, int k0) {
#pragma unroll
        for (int c = 0; c < 4; ++c) {
            int L = c * 1024 + t * 4;          // 128x32 tile, 4 halves per store
            int row = L >> 5, col = L & 31;
            v4h h;
            if (MODE == 0) {
                v4f x = *(const v4f*)(Xf + (m0 + row) * 1024 + k0 + col);
#pragma unroll
                for (int j = 0; j < 4; ++j) h[j] = (_Float16)x[j];
            } else {
                int m = m0 + row;
                int b = m >> 11, s = m & 2047;
                int k = k0 + col;
                int hh = k >> 6, dd = k & 63;  // 4 | 64 -> contiguous within a head
                h = *(const v4h*)(Xh + (((b << 4) + hh) * 2048 + s) * 64 + dd);
            }
            *(v4h*)(&sA[buf][row * 40 + col]) = h;
        }
    };
    auto fillB = [&](int buf, int k0) {
        // 4x4 block transpose: W[k][n] -> sB[n][k], all stores are b64
        int nb = t & 31, kb = t >> 5;
        int n = nb * 4, kk = kb * 4;
        v4f w[4];
#pragma unroll
        for (int j = 0; j < 4; ++j)
            w[j] = *(const v4f*)(W + (k0 + kk + j) * 1024 + n0 + n);
#pragma unroll
        for (int x = 0; x < 4; ++x) {
            v4h h;
#pragma unroll
            for (int j = 0; j < 4; ++j) h[j] = (_Float16)w[j][x];
            *(v4h*)(&sB[buf][(n + x) * 40 + kk]) = h;
        }
    };

    v8f acc[2][4] = {};

    fillA(0, 0);
    fillB(0, 0);
    __syncthreads();

    for (int kt = 0; kt < 32; ++kt) {
        const int k0  = kt * 32;
        const int cur = kt & 1;
        if (kt + 1 < 32) {                 // stage next tile into the other buffer
            fillA(cur ^ 1, k0 + 32);
            fillB(cur ^ 1, k0 + 32);
        }
        if (k0 + 64 < 1024) {              // pull K-tile kt+2 toward L2 (global_prefetch_b8)
            __builtin_prefetch(W + (k0 + 64 + (t >> 3)) * 1024 + n0 + (t & 7) * 16, 0, 1);
            if (MODE == 0)
                __builtin_prefetch(Xf + (m0 + t / 2) * 1024 + k0 + 64 + (t & 1) * 16, 0, 1);
        }

        v16h a0 = load_a_frag(sA[cur], wm * 32,      40, 0, lane);
        v16h a1 = load_a_frag(sA[cur], wm * 32 + 16, 40, 0, lane);
#pragma unroll
        for (int tn = 0; tn < 4; ++tn) {
            v16h b = load_bt_frag(sB[cur], wn * 64 + tn * 16, 40, 0, lane);
            acc[0][tn] = wmma_f16(a0, b, acc[0][tn]);
            acc[1][tn] = wmma_f16(a1, b, acc[1][tn]);
        }
        __syncthreads();                   // single barrier per K-step
    }

    // store C: lane -> col, VGPR g -> row g + 8*(lane>=16)
#pragma unroll
    for (int tm = 0; tm < 2; ++tm) {
#pragma unroll
        for (int tn = 0; tn < 4; ++tn) {
#pragma unroll
            for (int g = 0; g < 8; ++g) {
                int row = m0 + wm * 32 + tm * 16 + g + 8 * (lane >> 4);
                int col = n0 + wn * 64 + tn * 16 + (lane & 15);
                float val = acc[tm][tn][g] + bias[col];
                if (MODE == 0) {
                    int b = row >> 11, s = row & 2047;
                    int h = col >> 6, dd = col & 63;
                    Oh[(((b << 4) + h) * 2048 + s) * 64 + dd] = (_Float16)val;
                } else {
                    Of[row * 1024 + col] = val;
                }
            }
        }
    }
}

// ---------------- Kernel 2: strided-sparse attention (rows i%32 != 0) ----------------
// Per (b,h, 64-row q block): WMMA scores vs compact 64 strided K columns + exact diagonal.
// Q tile and strided-K tile are streamed by the Tensor Data Mover (HW-padded to ld=72)
// while all threads stage the transposed V tile; wave0 drains TENSORcnt, barrier publishes.
__global__ __launch_bounds__(128) void attn_strided(
    const _Float16* __restrict__ qh, const _Float16* __restrict__ kh,
    const _Float16* __restrict__ vh, _Float16* __restrict__ ctx) {

    __shared__ __align__(16) _Float16 sQ[64 * 72];   // Q rows (TDM), later reused for P
    __shared__ __align__(16) _Float16 sK[64 * 72];   // KcT [c][k] = kh[.., 32c, k] (TDM)
    __shared__ __align__(16) _Float16 sV[64 * 72];   // VcT [d][c] = vh[.., 32c, d]
    __shared__ float sS[64 * 65];                    // scores, ld=65 (bank-conflict pad)
    __shared__ float sDiag[64];
    __shared__ float sWd[64];

    const int t    = threadIdx.x;
    const int lane = t & 31;
    const int wave = t >> 5;              // 4 waves, one 16-row M tile each
    const int bh   = blockIdx.x >> 5;
    const int i0   = (blockIdx.x & 31) * 64;
    const int base = bh * HEADSZ;

    // TDM: 64x64 f16 tiles; row = 64 halves = 32 DW -> pad_interval code 4 (2^5=32 DW),
    // pad_amount code 3 (4 DW = 8 halves) -> LDS row pitch 72 halves.
    if (wave == 0) {
        tdm_load_tile_f16((unsigned)(unsigned long long)&sQ[0],
                          qh + base + i0 * 64, 64, 64, /*stride*/64, 4, 3);
        tdm_load_tile_f16((unsigned)(unsigned long long)&sK[0],
                          kh + base,           64, 64, /*stride*/32 * 64, 4, 3);
    }
    // stage compact-V transposed [d][c] via 4x4 blocks (all b64 LDS stores), overlapping TDM
#pragma unroll
    for (int p = 0; p < 2; ++p) {
        int bb = p * 128 + t;             // 256 4x4 blocks
        int c = (bb & 15) * 4, d = (bb >> 4) * 4;
        v4h r4[4];
#pragma unroll
        for (int j = 0; j < 4; ++j)
            r4[j] = *(const v4h*)(vh + base + (32 * (c + j)) * 64 + d);
#pragma unroll
        for (int x = 0; x < 4; ++x) {
            v4h hcol;
#pragma unroll
            for (int j = 0; j < 4; ++j) hcol[j] = r4[j][x];
            *(v4h*)(&sV[(d + x) * 72 + c]) = hcol;
        }
    }
    if (wave == 0) __builtin_amdgcn_s_wait_tensorcnt(0);
    __syncthreads();

    const int mbase = wave * 16;
    // scores = Q (16x64) x KcT (64 x 64 compact cols)
    {
        v8f acc[4] = {};
#pragma unroll
        for (int kc = 0; kc < 64; kc += 32) {
            v16h a = load_a_frag(sQ, mbase, 72, kc, lane);
#pragma unroll
            for (int tn = 0; tn < 4; ++tn) {
                v16h b = load_bt_frag(sK, tn * 16, 72, kc, lane);
                acc[tn] = wmma_f16(a, b, acc[tn]);
            }
        }
#pragma unroll
        for (int tn = 0; tn < 4; ++tn) {
#pragma unroll
            for (int g = 0; g < 8; ++g) {
                int rl = mbase + g + 8 * (lane >> 4);
                int i  = i0 + rl;
                int c  = tn * 16 + (lane & 15);
                int j  = 32 * c;
                float s = acc[tn][g] * 0.125f;           // 1/sqrt(64)
                if (j >= i) s = -1.0e30f;                // strided set: j%32==0 && j<i
                sS[rl * 65 + c] = s;
            }
        }
    }
    // exact diagonal term q_i . k_i
    if (t < 64) {
        int i = i0 + t;
        const v4h* kp = (const v4h*)(kh + base + i * 64);
        float d = 0.f;
#pragma unroll
        for (int k4 = 0; k4 < 16; ++k4) {
            v4h kv = kp[k4];
#pragma unroll
            for (int j = 0; j < 4; ++j)
                d += (float)sQ[t * 72 + k4 * 4 + j] * (float)kv[j];
        }
        sDiag[t] = d * 0.125f;
    }
    __syncthreads();

    // per-row softmax over {64 strided scores} U {diag}; write P (f16) over sQ
    if (t < 64) {
        float m = sDiag[t];
        for (int c = 0; c < 64; ++c) m = fmaxf(m, sS[t * 65 + c]);
        float sum = 0.f;
        for (int c = 0; c < 64; ++c) { float p = __expf(sS[t * 65 + c] - m); sS[t * 65 + c] = p; sum += p; }
        float pd  = __expf(sDiag[t] - m);
        float inv = 1.0f / (sum + pd);
        sWd[t] = pd * inv;
        for (int c = 0; c < 64; ++c) sQ[t * 72 + c] = (_Float16)(sS[t * 65 + c] * inv);
    }
    __syncthreads();

    // ctx = P (16x64) x VcT + wdiag * v_i ; skip rows i%32==0 (dense kernel owns them)
    {
        v8f acc[4] = {};
#pragma unroll
        for (int kc = 0; kc < 64; kc += 32) {
            v16h a = load_a_frag(sQ, mbase, 72, kc, lane);
#pragma unroll
            for (int tn = 0; tn < 4; ++tn) {
                v16h b = load_bt_frag(sV, tn * 16, 72, kc, lane);
                acc[tn] = wmma_f16(a, b, acc[tn]);
            }
        }
#pragma unroll
        for (int tn = 0; tn < 4; ++tn) {
#pragma unroll
            for (int g = 0; g < 8; ++g) {
                int rl = mbase + g + 8 * (lane >> 4);
                int i  = i0 + rl;
                if ((i & 31) != 0) {
                    int d = tn * 16 + (lane & 15);
                    float val = acc[tn][g] + sWd[rl] * (float)vh[base + i * 64 + d];
                    ctx[base + i * 64 + d] = (_Float16)val;
                }
            }
        }
    }
}

// ---------------- Kernel 3: dense "hub" rows (i%32 == 0), full 2048-col softmax ----------------
__global__ __launch_bounds__(256) void attn_dense(
    const _Float16* __restrict__ qh, const _Float16* __restrict__ kh,
    const _Float16* __restrict__ vh, _Float16* __restrict__ ctx) {

    __shared__ float qrow[64];
    __shared__ float sS[2048];
    __shared__ float red[256];
    __shared__ float part[256];

    const int t    = threadIdx.x;
    const int bh   = blockIdx.x >> 6;
    const int i    = (blockIdx.x & 63) * 32;   // hub row
    const int base = bh * HEADSZ;

    if (t < 64) qrow[t] = (float)qh[base + i * 64 + t];
    __syncthreads();

    // scores: row i attends { j >= i } U { j%32 == 0 }
    for (int j = t; j < 2048; j += 256) {
        float s = -1.0e30f;
        if (j >= i || (j & 31) == 0) {
            const v4h* kp = (const v4h*)(kh + base + j * 64);
            float d = 0.f;
#pragma unroll
            for (int k4 = 0; k4 < 16; ++k4) {
                v4h kv = kp[k4];
#pragma unroll
                for (int x = 0; x < 4; ++x) d += qrow[k4 * 4 + x] * (float)kv[x];
            }
            s = d * 0.125f;
        }
        sS[j] = s;
    }
    // block max
    float lm = -1.0e30f;
    for (int j = t; j < 2048; j += 256) lm = fmaxf(lm, sS[j]);
    red[t] = lm; __syncthreads();
    for (int off = 128; off > 0; off >>= 1) {
        if (t < off) red[t] = fmaxf(red[t], red[t + off]);
        __syncthreads();
    }
    float m = red[0]; __syncthreads();
    // exp + block sum
    float ls = 0.f;
    for (int j = t; j < 2048; j += 256) { float p = __expf(sS[j] - m); sS[j] = p; ls += p; }
    red[t] = ls; __syncthreads();
    for (int off = 128; off > 0; off >>= 1) {
        if (t < off) red[t] += red[t + off];
        __syncthreads();
    }
    float tinv = 1.0f / red[0];
    // weighted V: thread = (chunk, d)
    {
        int d = t & 63, ch = t >> 6;
        float a = 0.f;
        for (int jj = 0; jj < 512; ++jj) {
            int j = ch * 512 + jj;
            a += sS[j] * (float)vh[base + j * 64 + d];
        }
        part[t] = a;
    }
    __syncthreads();
    if (t < 64) {
        float v = (part[t] + part[t + 64] + part[t + 128] + part[t + 192]) * tinv;
        ctx[base + i * 64 + t] = (_Float16)v;
    }
}

// ---------------- launcher ----------------
extern "C" void kernel_launch(void* const* d_in, const int* in_sizes, int n_in,
                              void* d_out, int out_size, void* d_ws, size_t ws_size,
                              hipStream_t stream) {
    (void)in_sizes; (void)n_in; (void)out_size; (void)ws_size;
    const float* q  = (const float*)d_in[0];
    const float* k  = (const float*)d_in[1];
    const float* v  = (const float*)d_in[2];
    const float* Wq = (const float*)d_in[3];
    const float* bq = (const float*)d_in[4];
    const float* Wk = (const float*)d_in[5];
    const float* bk = (const float*)d_in[6];
    const float* Wv = (const float*)d_in[7];
    const float* bv = (const float*)d_in[8];
    const float* Wo = (const float*)d_in[9];
    const float* bo = (const float*)d_in[10];

    _Float16* ws  = (_Float16*)d_ws;
    _Float16* qh  = ws;
    _Float16* kh  = ws + (size_t)HELEMS;
    _Float16* vh  = ws + (size_t)2 * HELEMS;
    _Float16* ctx = ws + (size_t)3 * HELEMS;

    dim3 gg(1024 / 128, M_ / 128);   // (8, 32)
    gemm_wmma<0><<<gg, 256, 0, stream>>>(q, nullptr, Wq, bq, qh, nullptr);
    gemm_wmma<0><<<gg, 256, 0, stream>>>(k, nullptr, Wk, bk, kh, nullptr);
    gemm_wmma<0><<<gg, 256, 0, stream>>>(v, nullptr, Wv, bv, vh, nullptr);

    attn_strided<<<B_ * H_ * (S_ / 64), 128, 0, stream>>>(qh, kh, vh, ctx);
    attn_dense<<<B_ * H_ * 64, 256, 0, stream>>>(qh, kh, vh, ctx);

    gemm_wmma<1><<<gg, 256, 0, stream>>>(nullptr, ctx, Wo, bo, nullptr, (float*)d_out);
}
